// MambaVisionMixer_77197742178458
// MI455X (gfx1250) — matrix operand order repaired and checked by reading, not compiled, not run on hardware
//
#include <hip/hip_runtime.h>
#include <hip/hip_bf16.h>

// MambaVisionMixer for MI455X (gfx1250): bf16 WMMA GEMMs + chunk-parallel scan.
typedef __attribute__((ext_vector_type(16))) __bf16 v16bf;
typedef __attribute__((ext_vector_type(8)))  float  v8f;

#define BSZ   4
#define SEQ   4096
#define DM    256
#define NST   16
#define RDT   16
#define BL    (BSZ * SEQ)        // 16384 rows (b*l)
#define C3    (3 * DM)           // 768
#define XDBLC (RDT + 2 * NST)    // 48
#define NCH   (BSZ * DM)         // 1024 scan channels
#define CHK   64                 // chunk length
#define NCHK  (SEQ / CHK)        // 64 chunks per sequence

#define EPI_NONE  0
#define EPI_BIAS  1
#define EPI_ZGATE 2

// Pack 8 fp32 (two float4) into bf16 slots [base, base+8) of a v16bf fragment.
__device__ __forceinline__ void cvt8(v16bf& d, const int base, float4 lo, float4 hi) {
  d[base + 0] = (__bf16)lo.x; d[base + 1] = (__bf16)lo.y;
  d[base + 2] = (__bf16)lo.z; d[base + 3] = (__bf16)lo.w;
  d[base + 4] = (__bf16)hi.x; d[base + 5] = (__bf16)hi.y;
  d[base + 6] = (__bf16)hi.z; d[base + 7] = (__bf16)hi.w;
}

__device__ __forceinline__ float4 mul4(float4 a, float4 b) {
  return make_float4(a.x * b.x, a.y * b.y, a.z * b.z, a.w * b.w);
}

// ---------------------------------------------------------------------------
// Register-tiled GEMM: out[m,n] = sum_k Aop[m,k] * W[n,k] (+ epilogue).
// One wave computes an (MT*16) x (NT*16) block: MT*NT WMMA per k-step from
// MT+NT fragment loads. Fragments use contiguous k order (k = 16*half + s) in A and B.
// PROD: A element = A[m,k] * A2[m,k].  EPI: 0 none; 1 +bias[n];
// 2 z-gate: out = wcx[n]*(acc*Z[m,n]) + bcx[n].
// ---------------------------------------------------------------------------
template <int K, int LDA, int LDW, int LDC, int MT, int NT, int EPI, bool PROD>
__global__ __launch_bounds__(32)
void k_gemm(const float* __restrict__ A, const float* __restrict__ A2,
            const float* __restrict__ W,
            const float* __restrict__ aux0, const float* __restrict__ aux1,
            const float* __restrict__ aux2, float* __restrict__ out) {
  const int lane = threadIdx.x & 31;
  const int r = lane & 15, half = lane >> 4;
  const int mbase = blockIdx.x * (MT * 16);
  const int nbase = blockIdx.y * (NT * 16);
  const float4 z4 = make_float4(0.f, 0.f, 0.f, 0.f);

  const float* ap[MT];
  const float* ap2[MT];
  const float* wp[NT];
#pragma unroll
  for (int mt = 0; mt < MT; ++mt) {
    ap[mt]  = A  + (size_t)(mbase + mt * 16 + r) * LDA;
    ap2[mt] = PROD ? (A2 + (size_t)(mbase + mt * 16 + r) * LDA) : nullptr;
  }
#pragma unroll
  for (int nt = 0; nt < NT; ++nt)
    wp[nt] = W + (size_t)(nbase + nt * 16 + r) * LDW;  // r doubles as B column

  v8f acc[MT][NT];
#pragma unroll
  for (int mt = 0; mt < MT; ++mt)
#pragma unroll
    for (int nt = 0; nt < NT; ++nt) acc[mt][nt] = v8f{};

#pragma unroll
  for (int kk = 0; kk < K; kk += 32) {
    v16bf afr[MT], bfr[NT];
#pragma unroll
    for (int mt = 0; mt < MT; ++mt) {
      float4 lo = z4, hi = z4;
      int o = kk + 16 * half;
      if (o + 8 <= K) {
        lo = *(const float4*)(ap[mt] + o);
        hi = *(const float4*)(ap[mt] + o + 4);
        if (PROD) {
          lo = mul4(lo, *(const float4*)(ap2[mt] + o));
          hi = mul4(hi, *(const float4*)(ap2[mt] + o + 4));
        }
      }
      cvt8(afr[mt], 0, lo, hi);
      lo = z4; hi = z4;
      o = kk + 16 * half + 8;
      if (o + 8 <= K) {
        lo = *(const float4*)(ap[mt] + o);
        hi = *(const float4*)(ap[mt] + o + 4);
        if (PROD) {
          lo = mul4(lo, *(const float4*)(ap2[mt] + o));
          hi = mul4(hi, *(const float4*)(ap2[mt] + o + 4));
        }
      }
      cvt8(afr[mt], 8, lo, hi);
    }
#pragma unroll
    for (int nt = 0; nt < NT; ++nt) {
      float4 q0 = z4, q1 = z4, q2 = z4, q3 = z4;
      int o = kk + 16 * half;
      if (o + 16 <= K) {
        const float4* bp = (const float4*)(wp[nt] + o);
        q0 = bp[0]; q1 = bp[1]; q2 = bp[2]; q3 = bp[3];
      }
      cvt8(bfr[nt], 0, q0, q1);
      cvt8(bfr[nt], 8, q2, q3);
    }
#pragma unroll
    for (int mt = 0; mt < MT; ++mt)
#pragma unroll
      for (int nt = 0; nt < NT; ++nt)
        acc[mt][nt] = __builtin_amdgcn_wmma_f32_16x16x32_bf16(
            false, afr[mt], false, bfr[nt], (short)0, acc[mt][nt], false, false);
  }

  const int c = lane & 15;
#pragma unroll
  for (int mt = 0; mt < MT; ++mt) {
#pragma unroll
    for (int v = 0; v < 8; ++v) {
      const int m = mbase + mt * 16 + v + 8 * half;
#pragma unroll
      for (int nt = 0; nt < NT; ++nt) {
        const int n = nbase + nt * 16 + c;
        float res = acc[mt][nt][v];
        if constexpr (EPI == EPI_BIAS) {
          res += aux0[n];
        } else if constexpr (EPI == EPI_ZGATE) {
          res = aux1[n] * (res * aux0[(size_t)m * LDC + n]) + aux2[n];
        }
        out[(size_t)m * LDC + n] = res;
      }
    }
  }
}

// ---------------------------------------------------------------------------
// Depthwise conv (k=1 / k=3 / k=5, 'same', cross-correlation) + SiLU.
// ---------------------------------------------------------------------------
__device__ __forceinline__ float silu(float x) { return x / (1.f + __expf(-x)); }

__global__ void k_conv_silu(const float* __restrict__ xyz,
                            const float* __restrict__ wx, const float* __restrict__ bx,
                            const float* __restrict__ wy, const float* __restrict__ by,
                            const float* __restrict__ wz, const float* __restrict__ bz,
                            float* __restrict__ xo, float* __restrict__ yo,
                            float* __restrict__ zo) {
  int idx = blockIdx.x * blockDim.x + threadIdx.x;
  if (idx >= BL * DM) return;
  int m = idx >> 8;
  int d = idx & (DM - 1);
  int l = m & (SEQ - 1);

  float sx = wx[d] * xyz[(size_t)m * C3 + d] + bx[d];
  float sy = by[d];
#pragma unroll
  for (int t = 0; t < 3; ++t) {
    int lp = l - 1 + t;
    if (lp >= 0 && lp < SEQ)
      sy += wy[d * 3 + t] * xyz[(size_t)(m - 1 + t) * C3 + DM + d];
  }
  float sz = bz[d];
#pragma unroll
  for (int t = 0; t < 5; ++t) {
    int lp = l - 2 + t;
    if (lp >= 0 && lp < SEQ)
      sz += wz[d * 5 + t] * xyz[(size_t)(m - 2 + t) * C3 + 2 * DM + d];
  }
  xo[idx] = silu(sx);
  yo[idx] = silu(sy);
  zo[idx] = silu(sz);
}

// ---------------------------------------------------------------------------
// Chunk-parallel selective scan.  h_{l+1} = a_l.h_l + b_l (elementwise, 16-wide)
// is affine, so a chunk of 64 steps composes to (Ac = prod a, Hc = run from 0).
// Phase 1: per (chunk, channel, n) lane, compute Ac/Hc.          [1M lanes]
// Phase 2: per (channel, n) lane, serial scan over 64 chunks -> h0 per chunk.
// Phase 3: re-run each chunk from exact h0, emit y via shfl reduction.
// Serial depth drops 4096 -> ~3x64.
// ---------------------------------------------------------------------------
__device__ __forceinline__ void scan_step(float m_dt, float m_u, float Bv,
                                          float Aval, float bias, float& h,
                                          float& aprod, bool track_a) {
  float dtv = m_dt + bias;                                // bias added twice (faithful)
  float sp  = (dtv > 20.f) ? dtv : log1pf(__expf(dtv));   // softplus
  float a   = __expf(sp * Aval);
  h = a * h + sp * Bv * m_u;
  if (track_a) aprod *= a;
}

__global__ __launch_bounds__(256)
void k_scan_ph1(const float* __restrict__ dtbuf, const float* __restrict__ xcv,
                const float* __restrict__ xdbl, const float* __restrict__ A_log,
                const float* __restrict__ dt_b,
                float* __restrict__ Ac, float* __restrict__ Hc) {
  const int tid = threadIdx.x;
  const int c   = blockIdx.x & (NCHK - 1);                 // chunk id
  const int ch  = (blockIdx.x >> 6) * 16 + (tid >> 4);     // channel
  const int n   = tid & 15;
  const int b   = ch >> 8;
  const int d   = ch & (DM - 1);

  const float Aval = -__expf(A_log[d * NST + n]);
  const float bias = dt_b[d];
  const int   m0   = b * SEQ + c * CHK;

  float h = 0.f, aprod = 1.f;
  for (int i = 0; i < CHK; ++i) {
    const int m = m0 + i;
    if (i + 4 < CHK) {
      __builtin_prefetch(&dtbuf[(size_t)(m + 4) * DM + d], 0, 1);
      __builtin_prefetch(&xdbl[(size_t)(m + 4) * XDBLC + n], 0, 1);
    }
    scan_step(dtbuf[(size_t)m * DM + d], xcv[(size_t)m * DM + d],
              xdbl[(size_t)m * XDBLC + RDT + n], Aval, bias, h, aprod, true);
  }
  const size_t idx = (size_t)c * (NCH * NST) + (size_t)ch * NST + n;
  Ac[idx] = aprod;
  Hc[idx] = h;
}

__global__ __launch_bounds__(256)
void k_scan_ph2(const float* __restrict__ Ac, const float* __restrict__ Hc,
                float* __restrict__ h0) {
  const int lane = blockIdx.x * 256 + threadIdx.x;         // (ch, n) in [0, 16384)
  float h = 0.f;
  for (int c = 0; c < NCHK; ++c) {
    const size_t idx = (size_t)c * (NCH * NST) + lane;
    h0[idx] = h;                                           // state entering chunk c
    h = Ac[idx] * h + Hc[idx];
  }
}

__global__ __launch_bounds__(256)
void k_scan_ph3(const float* __restrict__ dtbuf, const float* __restrict__ xcv,
                const float* __restrict__ xdbl, const float* __restrict__ A_log,
                const float* __restrict__ dt_b, const float* __restrict__ Dvec,
                const float* __restrict__ h0, float* __restrict__ pbuf) {
  const int tid = threadIdx.x;
  const int c   = blockIdx.x & (NCHK - 1);
  const int ch  = (blockIdx.x >> 6) * 16 + (tid >> 4);
  const int n   = tid & 15;
  const int b   = ch >> 8;
  const int d   = ch & (DM - 1);

  const float Aval = -__expf(A_log[d * NST + n]);
  const float bias = dt_b[d];
  const float Dv   = Dvec[d];
  const int   m0   = b * SEQ + c * CHK;

  float h = h0[(size_t)c * (NCH * NST) + (size_t)ch * NST + n];
  float dummy = 1.f;
  for (int i = 0; i < CHK; ++i) {
    const int m = m0 + i;
    if (i + 4 < CHK) {
      __builtin_prefetch(&dtbuf[(size_t)(m + 4) * DM + d], 0, 1);
      __builtin_prefetch(&xdbl[(size_t)(m + 4) * XDBLC + n], 0, 1);
    }
    const float u = xcv[(size_t)m * DM + d];
    scan_step(dtbuf[(size_t)m * DM + d], u,
              xdbl[(size_t)m * XDBLC + RDT + n], Aval, bias, h, dummy, false);
    float yp = h * xdbl[(size_t)m * XDBLC + RDT + NST + n];   // h * C
#pragma unroll
    for (int off = 8; off >= 1; off >>= 1) yp += __shfl_xor(yp, off, 32);
    if (n == 0) pbuf[(size_t)m * DM + d] = yp + Dv * u;
  }
}

// ---------------------------------------------------------------------------
extern "C" void kernel_launch(void* const* d_in, const int* in_sizes, int n_in,
                              void* d_out, int out_size, void* d_ws, size_t ws_size,
                              hipStream_t stream) {
  (void)in_sizes; (void)n_in; (void)out_size; (void)ws_size;
  const float* hs        = (const float*)d_in[0];   // (4,4096,256)
  const float* in_proj_w = (const float*)d_in[1];   // (768,256)
  const float* x_proj_w  = (const float*)d_in[2];   // (48,256)
  const float* dt_proj_w = (const float*)d_in[3];   // (256,16)
  const float* dt_proj_b = (const float*)d_in[4];   // (256,)
  const float* A_log     = (const float*)d_in[5];   // (256,16)
  const float* Dvec      = (const float*)d_in[6];   // (256,)
  const float* out_w     = (const float*)d_in[7];   // (256,256)
  const float* conv_x_w  = (const float*)d_in[8];
  const float* conv_x_b  = (const float*)d_in[9];
  const float* conv_y_w  = (const float*)d_in[10];
  const float* conv_y_b  = (const float*)d_in[11];
  const float* conv_z_w  = (const float*)d_in[12];
  const float* conv_z_b  = (const float*)d_in[13];
  float* outp = (float*)d_out;

  char* ws = (char*)d_ws;
  size_t off = 0;
  auto alloc = [&](size_t bytes) -> float* {
    float* p = (float*)(ws + off);
    off += (bytes + 255) & ~(size_t)255;
    return p;
  };
  float* xyz_raw = alloc((size_t)BL * C3 * 4);         // 48 MB
  float* xcv     = alloc((size_t)BL * DM * 4);         // 16 MB
  float* ycv     = alloc((size_t)BL * DM * 4);
  float* zcv     = alloc((size_t)BL * DM * 4);
  float* xdbl    = alloc((size_t)BL * XDBLC * 4);      // 3 MB
  float* dtbuf   = alloc((size_t)BL * DM * 4);
  float* pbuf    = alloc((size_t)BL * DM * 4);
  float* Ac      = alloc((size_t)NCHK * NCH * NST * 4);  // 4 MB
  float* Hc      = alloc((size_t)NCHK * NCH * NST * 4);
  float* h0      = alloc((size_t)NCHK * NCH * NST * 4);

  // 1) in_proj GEMM -> xyz_raw (16384 x 768); 32x64 per wave
  k_gemm<DM, DM, DM, C3, 2, 4, EPI_NONE, false>
      <<<dim3(BL / 32, C3 / 64), 32, 0, stream>>>(
          hs, nullptr, in_proj_w, nullptr, nullptr, nullptr, xyz_raw);

  // 2) depthwise convs + SiLU -> x, y, z
  k_conv_silu<<<(BL * DM) / 256, 256, 0, stream>>>(
      xyz_raw, conv_x_w, conv_x_b, conv_y_w, conv_y_b, conv_z_w, conv_z_b,
      xcv, ycv, zcv);

  // 3) x_proj GEMM -> xdbl = [dt_r | B | C] (16384 x 48); 16x48 per wave
  k_gemm<DM, DM, DM, XDBLC, 1, 3, EPI_NONE, false>
      <<<dim3(BL / 16, 1), 32, 0, stream>>>(
          xcv, nullptr, x_proj_w, nullptr, nullptr, nullptr, xdbl);

  // 4) dt GEMM (K=16, zero-padded fragments) + bias -> dtbuf (16384 x 256)
  k_gemm<RDT, XDBLC, RDT, DM, 1, 4, EPI_BIAS, false>
      <<<dim3(BL / 16, DM / 64), 32, 0, stream>>>(
          xdbl, nullptr, dt_proj_w, dt_proj_b, nullptr, nullptr, dtbuf);

  // 5) chunk-parallel selective scan -> pbuf
  k_scan_ph1<<<NCHK * (NCH / 16), 256, 0, stream>>>(
      dtbuf, xcv, xdbl, A_log, dt_proj_b, Ac, Hc);
  k_scan_ph2<<<(NCH * NST) / 256, 256, 0, stream>>>(Ac, Hc, h0);
  k_scan_ph3<<<NCHK * (NCH / 16), 256, 0, stream>>>(
      dtbuf, xcv, xdbl, A_log, dt_proj_b, Dvec, h0, pbuf);

  // 6) out_proj: l1 = (p*y) @ Wout^T, gated by z, final k=1 conv -> d_out
  k_gemm<DM, DM, DM, DM, 2, 4, EPI_ZGATE, true>
      <<<dim3(BL / 32, DM / 64), 32, 0, stream>>>(
          pbuf, ycv, out_w, zcv, conv_x_w, conv_x_b, outp);
}
